// S4sequence_17420387352582
// MI455X (gfx1250) — compile-verified, matrix operand order checked by reading, not run on hardware
//
#include <hip/hip_runtime.h>
#include <hip/hip_bf16.h>
#include <math.h>

// Shapes (compile-time, from the reference)
#define Bb   8
#define Ll   2048
#define Hh   512
#define Nn   16
#define Rr   32
#define PROJ 64            // R + 2N
#define Mm   (Bb * Ll)     // 16384 rows of the flattened [b,l] dim

typedef __attribute__((ext_vector_type(16))) __bf16 v16bf;
typedef __attribute__((ext_vector_type(8)))  __bf16 v8bf;
typedef __attribute__((ext_vector_type(8)))  float  v8f;
typedef __attribute__((ext_vector_type(8)))  float  f8_t;

// ---------------------------------------------------------------------------
// CDNA5 WMMA fragment loaders. A is 16x32 (MxK) bf16, B is 32x16 (KxN) bf16.
// Per the ISA layout: lane L (0..31): row/col = L&15, k-offset = (L>>4)*8,
// elements 0..7 = [k+koff .. k+koff+7], elements 8..15 = [k+koff+16 ..].
// All three GEMMs are X @ W^T with W row-major [n][k], so A rows and B columns
// load with the identical per-lane pattern.
// ---------------------------------------------------------------------------
__device__ __forceinline__ v16bf load_tile_bf16(const __bf16* base, int ld,
                                                int row0, int k) {
    int lane = threadIdx.x & 31;
    int r    = row0 + (lane & 15);
    int koff = k + ((lane >> 4) << 3);
    const v8bf* p0 = reinterpret_cast<const v8bf*>(base + (size_t)r * ld + koff);
    const v8bf* p1 = reinterpret_cast<const v8bf*>(base + (size_t)r * ld + koff + 16);
    union { v16bf v; v8bf h[2]; } u;
    u.h[0] = *p0;
    u.h[1] = *p1;
    return u.v;
}

__device__ __forceinline__ v16bf load_tile_f32cvt(const float* base, int ld,
                                                  int row0, int k) {
    int lane = threadIdx.x & 31;
    int r    = row0 + (lane & 15);
    int koff = k + ((lane >> 4) << 3);
    const f8_t* p0 = reinterpret_cast<const f8_t*>(base + (size_t)r * ld + koff);
    const f8_t* p1 = reinterpret_cast<const f8_t*>(base + (size_t)r * ld + koff + 16);
    union { v16bf v; v8bf h[2]; } u;
    u.h[0] = __builtin_convertvector(*p0, v8bf);
    u.h[1] = __builtin_convertvector(*p1, v8bf);
    return u.v;
}

// ---------------------------------------------------------------------------
// 0) zero the LN accumulator slots
// ---------------------------------------------------------------------------
__global__ void zero_sums_kernel(float* sums) {
    if (threadIdx.x < 2 * Bb) sums[threadIdx.x] = 0.0f;
}

// ---------------------------------------------------------------------------
// 1) convert the three weight matrices to bf16 (done once per launch; tiny)
// ---------------------------------------------------------------------------
__global__ void cvt_weights_kernel(const float* wx, const float* wdt, const float* wout,
                                   __bf16* wxb, __bf16* wdtb, __bf16* woutb) {
    const int n0 = PROJ * Hh;          // 32768
    const int n1 = n0 + Hh * Rr;       // 49152
    const int nt = n1 + Hh * Hh;       // 311296
    for (int i = blockIdx.x * blockDim.x + threadIdx.x; i < nt;
         i += gridDim.x * blockDim.x) {
        if (i < n0)       wxb[i]        = (__bf16)wx[i];
        else if (i < n1)  wdtb[i - n0]  = (__bf16)wdt[i - n0];
        else              woutb[i - n1] = (__bf16)wout[i - n1];
    }
}

// ---------------------------------------------------------------------------
// 2) LayerNorm reduction: per batch, sum and sum-of-squares over L*H elements
// grid = Bb * 64 blocks of 256 threads; each block owns a 16384-elem slice.
// ---------------------------------------------------------------------------
__global__ __launch_bounds__(256) void ln_reduce_kernel(const float* x, float* sums) {
    int b     = blockIdx.x >> 6;
    int slice = blockIdx.x & 63;
    const int per_block = (Ll * Hh) / 64;    // 16384
    const float* xb = x + (size_t)b * (Ll * Hh) + (size_t)slice * per_block;

    float s1 = 0.0f, s2 = 0.0f;
    for (int i = threadIdx.x; i < per_block; i += 256) {
        float v = xb[i];
        s1 += v;
        s2 += v * v;
    }
    __shared__ float sh1[256];
    __shared__ float sh2[256];
    sh1[threadIdx.x] = s1;
    sh2[threadIdx.x] = s2;
    __syncthreads();
    for (int off = 128; off > 0; off >>= 1) {
        if ((int)threadIdx.x < off) {
            sh1[threadIdx.x] += sh1[threadIdx.x + off];
            sh2[threadIdx.x] += sh2[threadIdx.x + off];
        }
        __syncthreads();
    }
    if (threadIdx.x == 0) {
        atomicAdd(&sums[2 * b + 0], sh1[0]);
        atomicAdd(&sums[2 * b + 1], sh2[0]);
    }
}

// ---------------------------------------------------------------------------
// 3) LayerNorm apply: xn = (x - mu) * rsqrt(var + 1e-5); store f32 (scan u)
//    and bf16 (WMMA A operand).  8M elements, one thread each.
// ---------------------------------------------------------------------------
__global__ __launch_bounds__(256) void ln_apply_kernel(const float* x, const float* sums,
                                                       float* xnf, __bf16* xnb) {
    size_t idx = (size_t)blockIdx.x * blockDim.x + threadIdx.x;   // < 8*2^20
    int b = (int)(idx >> 20);                                     // L*H = 2^20
    const float invN = 1.0f / (float)(Ll * Hh);
    float mu  = sums[2 * b + 0] * invN;
    float var = sums[2 * b + 1] * invN - mu * mu;
    float inv = rsqrtf(var + 1e-5f);
    float v = (x[idx] - mu) * inv;
    xnf[idx] = v;
    xnb[idx] = (__bf16)v;
}

// ---------------------------------------------------------------------------
// 4) GEMM1: x_dbl[m, 0:64] = xn[m, :] @ x_proj_w[k_out, :]^T  (K = 512)
// One wave -> 16 rows x full 64 cols (4 accumulators). 1024 waves total.
// ---------------------------------------------------------------------------
__global__ __launch_bounds__(256) void gemm_xdbl_kernel(const __bf16* xn,
                                                        const __bf16* wx,
                                                        float* xdbl) {
    int wave = (int)((blockIdx.x * blockDim.x + threadIdx.x) >> 5);
    int row0 = wave * 16;

    v8f acc[4] = {};
    for (int k = 0; k < Hh; k += 32) {
        v16bf a = load_tile_bf16(xn, Hh, row0, k);
#pragma unroll
        for (int c = 0; c < 4; ++c) {
            v16bf bm = load_tile_bf16(wx, Hh, c * 16, k);
            acc[c] = __builtin_amdgcn_wmma_f32_16x16x32_bf16(
                false, a, false, bm, (short)0, acc[c], false, false);
        }
    }
    int lane  = threadIdx.x & 31;
    int mbase = row0 + ((lane >> 4) << 3);
    int ncol  = lane & 15;
#pragma unroll
    for (int c = 0; c < 4; ++c)
#pragma unroll
        for (int r = 0; r < 8; ++r)
            xdbl[(size_t)(mbase + r) * PROJ + c * 16 + ncol] = acc[c][r];
}

// ---------------------------------------------------------------------------
// 5) GEMM2: delta[m, d] = softplus( dt_r[m, :] @ dt_proj_w[d, :]^T + b[d] )
// K = 32 -> a single WMMA k-step. Wave -> 16 rows x 64 cols. 8192 waves.
// ---------------------------------------------------------------------------
__global__ __launch_bounds__(256) void gemm_delta_kernel(const float* xdbl,
                                                         const __bf16* wdt,
                                                         const float* dtb,
                                                         float* delta) {
    int wave = (int)((blockIdx.x * blockDim.x + threadIdx.x) >> 5);
    int row0 = (wave >> 3) * 16;
    int col0 = (wave & 7) * 64;

    v8f acc[4] = {};
    v16bf a = load_tile_f32cvt(xdbl, PROJ, row0, 0);   // dt_r = cols 0..31
#pragma unroll
    for (int c = 0; c < 4; ++c) {
        v16bf bm = load_tile_bf16(wdt, Rr, col0 + c * 16, 0);
        acc[c] = __builtin_amdgcn_wmma_f32_16x16x32_bf16(
            false, a, false, bm, (short)0, acc[c], false, false);
    }
    int lane  = threadIdx.x & 31;
    int mbase = row0 + ((lane >> 4) << 3);
    int ncol  = lane & 15;
#pragma unroll
    for (int c = 0; c < 4; ++c)
#pragma unroll
        for (int r = 0; r < 8; ++r) {
            int o = col0 + c * 16 + ncol;
            float v = acc[c][r] + dtb[o];
            v = (v > 20.0f) ? v : log1pf(__expf(v));   // softplus
            delta[(size_t)(mbase + r) * Hh + o] = v;
        }
}

// ---------------------------------------------------------------------------
// 6) Selective scan + D-skip + exact GeLU -> bf16 matrix for the output GEMM.
// One thread per (b, d, n); 16 n-lanes per half-wave reduce y with shfl_xor.
// ---------------------------------------------------------------------------
__global__ __launch_bounds__(256) void scan_kernel(const float* __restrict__ delta,
                                                   const float* __restrict__ xn,
                                                   const float* __restrict__ xdbl,
                                                   const float* __restrict__ A_log,
                                                   const float* __restrict__ Dv,
                                                   __bf16* __restrict__ yg) {
    int t = blockIdx.x * blockDim.x + threadIdx.x;   // 0 .. 65535
    int n = t & (Nn - 1);
    int d = (t >> 4) & (Hh - 1);
    int b = t >> 13;

    float Adn = -__expf(A_log[d * Nn + n]);
    float Dd  = Dv[d];

    const float* dptr  = delta + (size_t)b * Ll * Hh + d;
    const float* uptr  = xn    + (size_t)b * Ll * Hh + d;
    const float* bcptr = xdbl  + (size_t)b * Ll * PROJ;
    __bf16* yptr = yg + (size_t)b * Ll * Hh + d;

    float state = 0.0f;
    for (int l = 0; l < Ll; ++l) {
        float dv = dptr[(size_t)l * Hh];                 // broadcast in 16-group
        float uv = uptr[(size_t)l * Hh];                 // broadcast in 16-group
        float Bv = bcptr[l * PROJ + Rr + n];             // coalesced over n
        float Cv = bcptr[l * PROJ + Rr + Nn + n];        // coalesced over n

        state = state * __expf(dv * Adn) + dv * Bv * uv;
        float y = state * Cv;
        // reduce over the 16 n-lanes (xor masks < 16 stay inside the group)
        y += __shfl_xor(y, 1, 32);
        y += __shfl_xor(y, 2, 32);
        y += __shfl_xor(y, 4, 32);
        y += __shfl_xor(y, 8, 32);
        if (n == 0) {
            float yf = y + uv * Dd;
            float g  = 0.5f * yf * (1.0f + erff(yf * 0.70710678118654752f));
            yptr[(size_t)l * Hh] = (__bf16)g;
        }
    }
}

// ---------------------------------------------------------------------------
// 7) GEMM3: out[m, o] = gelu_y[m, :] @ out1_w[o, :]^T + b[o] + x[m, o]
// K = 512. Wave -> 16 rows x 64 cols (4 accs, A reused). 8192 waves.
// ---------------------------------------------------------------------------
__global__ __launch_bounds__(256) void gemm_out_kernel(const __bf16* yg,
                                                       const __bf16* wout,
                                                       const float* bias,
                                                       const float* xskip,
                                                       float* out) {
    int wave = (int)((blockIdx.x * blockDim.x + threadIdx.x) >> 5);
    int row0 = (wave >> 3) * 16;
    int col0 = (wave & 7) * 64;

    v8f acc[4] = {};
    for (int k = 0; k < Hh; k += 32) {
        v16bf a = load_tile_bf16(yg, Hh, row0, k);
#pragma unroll
        for (int c = 0; c < 4; ++c) {
            v16bf bm = load_tile_bf16(wout, Hh, col0 + c * 16, k);
            acc[c] = __builtin_amdgcn_wmma_f32_16x16x32_bf16(
                false, a, false, bm, (short)0, acc[c], false, false);
        }
    }
    int lane  = threadIdx.x & 31;
    int mbase = row0 + ((lane >> 4) << 3);
    int ncol  = lane & 15;
#pragma unroll
    for (int c = 0; c < 4; ++c)
#pragma unroll
        for (int r = 0; r < 8; ++r) {
            int o = col0 + c * 16 + ncol;
            size_t mi = (size_t)(mbase + r) * Hh + o;
            out[mi] = acc[c][r] + bias[o] + xskip[mi];
        }
}

// ---------------------------------------------------------------------------
// launcher
// ---------------------------------------------------------------------------
extern "C" void kernel_launch(void* const* d_in, const int* in_sizes, int n_in,
                              void* d_out, int out_size, void* d_ws, size_t ws_size,
                              hipStream_t stream) {
    (void)in_sizes; (void)n_in; (void)out_size; (void)ws_size;

    const float* x     = (const float*)d_in[0];   // [8,2048,512]
    const float* wx    = (const float*)d_in[1];   // [64,512]
    const float* wdt   = (const float*)d_in[2];   // [512,32]
    const float* dtb   = (const float*)d_in[3];   // [512]
    const float* A_log = (const float*)d_in[4];   // [512,16]
    const float* Dv    = (const float*)d_in[5];   // [512]
    const float* wout  = (const float*)d_in[6];   // [512,512]
    const float* outb  = (const float*)d_in[7];   // [512]
    float* out = (float*)d_out;

    // carve workspace (256B aligned chunks)
    char* w = (char*)d_ws;
    auto carve = [&](size_t bytes) {
        char* p = w;
        w += (bytes + 255) & ~(size_t)255;
        return p;
    };
    float*  sums  = (float*)carve(2 * Bb * sizeof(float));
    float*  xnf   = (float*)carve(sizeof(float) * (size_t)Mm * Hh);      // 33.5 MB
    __bf16* xnb   = (__bf16*)carve(2ull * Mm * Hh);                      // 16.8 MB
    float*  xdbl  = (float*)carve(sizeof(float) * (size_t)Mm * PROJ);    //  4.2 MB
    float*  delta = (float*)carve(sizeof(float) * (size_t)Mm * Hh);      // 33.5 MB
    __bf16* yg    = (__bf16*)carve(2ull * Mm * Hh);                      // 16.8 MB
    __bf16* wxb   = (__bf16*)carve(2ull * PROJ * Hh);
    __bf16* wdtb  = (__bf16*)carve(2ull * Hh * Rr);
    __bf16* woutb = (__bf16*)carve(2ull * Hh * Hh);

    zero_sums_kernel<<<1, 64, 0, stream>>>(sums);
    cvt_weights_kernel<<<256, 256, 0, stream>>>(wx, wdt, wout, wxb, wdtb, woutb);
    ln_reduce_kernel<<<Bb * 64, 256, 0, stream>>>(x, sums);
    ln_apply_kernel<<<(Mm * Hh) / 256, 256, 0, stream>>>(x, sums, xnf, xnb);
    gemm_xdbl_kernel<<<(Mm / 16) / 8, 256, 0, stream>>>(xnb, wxb, xdbl);          // 1024 waves
    gemm_delta_kernel<<<((Mm / 16) * (Hh / 64)) / 8, 256, 0, stream>>>(xdbl, wdtb, dtb, delta); // 8192 waves
    scan_kernel<<<(Bb * Hh * Nn) / 256, 256, 0, stream>>>(delta, xnf, xdbl, A_log, Dv, yg);
    gemm_out_kernel<<<((Mm / 16) * (Hh / 64)) / 8, 256, 0, stream>>>(yg, woutb, outb, x, out);
}
